// FastformerEncoder_15874199126560
// MI455X (gfx1250) — compile-verified, matrix operand order checked by reading, not run on hardware
//
#include <hip/hip_runtime.h>
#include <hip/hip_bf16.h>
#include <math.h>

typedef __attribute__((ext_vector_type(16))) __bf16 v16bf;
typedef __attribute__((ext_vector_type(8)))  float  v8f;

constexpr int Bc    = 8;
constexpr int Nseq  = 512;
constexpr int Hc    = 768;
constexpr int Lc    = 6;
constexpr int Ic    = 3072;
constexpr int HEADS = 12;
constexpr int Dd    = 64;
constexpr int Gg    = 2;
constexpr int HG    = HEADS * Gg;     // 24
constexpr int Mrows = Bc * Nseq;      // 4096

constexpr int BM = 128, BN = 128, BKK = 64;

// ---- gfx1250 async global->LDS DMA (ASYNCcnt-tracked) ----------------------
#define WAIT_ASYNC(n) asm volatile("s_wait_asynccnt %0" :: "i"(n) : "memory")

__device__ __forceinline__ unsigned lds_addr32(const void* p) {
    return (unsigned)(size_t)p;     // low 32 bits of shared-aperture address
}
__device__ __forceinline__ void async_ld16(unsigned lds, const void* g) {
    asm volatile("global_load_async_to_lds_b128 %0, %1, off"
                 :: "v"(lds), "v"(g) : "memory");
}

// ---------------------------------------------------------------------------
// bf16 WMMA GEMM:  C[M,N] = A[M,K] @ W[K,N] + bias (+ epilogue)
//   A   : bf16 [M,K] row-major ; Wtr : bf16 [N,K] (pre-transposed weights)
// Block: 128 threads = 4 waves; block tile 128x128; wave tile 64x64;
// K-stage 64, double-buffered LDS filled by async DMA.
// EPI: 0 bias, 1 bias+res, 2 bias+GELU, 3 bias+tanh.  WB: write bf16 mirror.
// ---------------------------------------------------------------------------
template <int EPI, bool WB, int K>
__global__ __launch_bounds__(128)
void gemm_wmma(const __bf16* __restrict__ A, const __bf16* __restrict__ Wtr,
               const float* __restrict__ bias, const float* __restrict__ res,
               float* __restrict__ C, __bf16* __restrict__ Cbf, int N)
{
    __shared__ alignas(32) __bf16 As[2][BM][BKK];   // [stage][m][k]  16KB each
    __shared__ alignas(32) __bf16 Bs[2][BN][BKK];   // [stage][n][k]

    const int tid = threadIdx.x;
    const int m0  = blockIdx.y * BM;
    const int n0  = blockIdx.x * BN;

    const int w     = tid >> 5;       // wave 0..3
    const int lane  = tid & 31;
    const int wm    = w >> 1;         // 0..1 (64-row group)
    const int wn    = w & 1;          // 0..1 (64-col group)
    const int lrow  = lane & 15;
    const int khalf = lane >> 4;      // 0: K0..15, 1: K16..31 within a 32-chunk

    v8f acc[4][4];
#pragma unroll
    for (int mi = 0; mi < 4; ++mi)
#pragma unroll
        for (int ni = 0; ni < 4; ++ni)
            acc[mi][ni] = (v8f){0.f,0.f,0.f,0.f,0.f,0.f,0.f,0.f};

    // async fill: each thread owns one 128B tile row (8 x 16B chunks) per tile
    const int rowT = tid;                               // 0..127
    const char* Abase = (const char*)(A   + (size_t)(m0 + rowT) * K);
    const char* Bbase = (const char*)(Wtr + (size_t)(n0 + rowT) * K);

    auto issue = [&](int k0, int st) {
        const char* ga = Abase + (size_t)k0 * 2;
        const char* gb = Bbase + (size_t)k0 * 2;
        unsigned    la = lds_addr32(&As[st][rowT][0]);
        unsigned    lb = lds_addr32(&Bs[st][rowT][0]);
#pragma unroll
        for (int c = 0; c < 8; ++c) {
            async_ld16(la + c * 16, ga + c * 16);
            async_ld16(lb + c * 16, gb + c * 16);
        }
    };

    auto compute = [&](int st) {
#pragma unroll
        for (int j = 0; j < 2; ++j) {                   // two 32-deep WMMA steps
            v16bf afr[4], bfr[4];
#pragma unroll
            for (int mi = 0; mi < 4; ++mi)
                afr[mi] = *(const v16bf*)&As[st][wm * 64 + mi * 16 + lrow][j * 32 + khalf * 16];
#pragma unroll
            for (int ni = 0; ni < 4; ++ni)
                bfr[ni] = *(const v16bf*)&Bs[st][wn * 64 + ni * 16 + lrow][j * 32 + khalf * 16];
#pragma unroll
            for (int mi = 0; mi < 4; ++mi)
#pragma unroll
                for (int ni = 0; ni < 4; ++ni)
                    acc[mi][ni] = __builtin_amdgcn_wmma_f32_16x16x32_bf16(
                        false, afr[mi], false, bfr[ni],
                        (short)0, acc[mi][ni], false, false);
        }
    };

    constexpr int NSTAGE = K / BKK;
    issue(0, 0);
    int s = 0;
#pragma unroll 1
    for (int st = 0; st < NSTAGE - 1; ++st, s ^= 1) {   // branch-free steady state
        issue((st + 1) * BKK, s ^ 1);
        WAIT_ASYNC(16);                                 // retire current stage only
        __syncthreads();
        compute(s);
        __syncthreads();                                // reads done before reuse
    }
    WAIT_ASYNC(0);                                      // peeled last stage
    __syncthreads();
    compute(s);

    // Epilogue: VGPR r -> M=r (lanes 0-15) / r+8 (lanes 16-31); N = lane%16
#pragma unroll
    for (int mi = 0; mi < 4; ++mi) {
#pragma unroll
        for (int ni = 0; ni < 4; ++ni) {
            const int col  = n0 + wn * 64 + ni * 16 + lrow;
            const float bc = bias[col];
#pragma unroll
            for (int r = 0; r < 8; ++r) {
                const int row = m0 + wm * 64 + mi * 16 + khalf * 8 + r;
                float v = acc[mi][ni][r] + bc;
                if (EPI == 1) v += res[(size_t)row * N + col];
                if (EPI == 2) v = 0.5f * v * (1.0f + erff(v * 0.70710678118654752f));
                if (EPI == 3) v = tanhf(v);
                C[(size_t)row * N + col] = v;
                if (WB) Cbf[(size_t)row * N + col] = (__bf16)v;
            }
        }
    }
}

// ---------------------------------------------------------------------------
// LDS-tiled weight convert+transpose: Wt[n*K + k] = (bf16) W[k*N + n]
// grid = (N/32, K/32), 256 threads (32 cols x 8 row-groups of 4)
// ---------------------------------------------------------------------------
__global__ __launch_bounds__(256)
void cvt_w_kernel(const float* __restrict__ W, __bf16* __restrict__ Wt, int K, int N)
{
    __shared__ float t[32][33];
    const int c  = threadIdx.x & 31;
    const int r0 = (threadIdx.x >> 5) * 4;
    const int n0 = blockIdx.x * 32;
    const int k0 = blockIdx.y * 32;
#pragma unroll
    for (int i = 0; i < 4; ++i)
        t[r0 + i][c] = W[(size_t)(k0 + r0 + i) * N + n0 + c];   // coalesced read
    __syncthreads();
#pragma unroll
    for (int i = 0; i < 4; ++i)
        Wt[(size_t)(n0 + r0 + i) * K + k0 + c] = (__bf16)t[c][r0 + i]; // coalesced write
}

// ---------------------------------------------------------------------------
// LayerNorm over H=768 per row; optional positional add; optional bf16 mirror.
// ---------------------------------------------------------------------------
__global__ __launch_bounds__(256)
void ln_kernel(const float* __restrict__ X, const float* __restrict__ pos,
               const float* __restrict__ gam, const float* __restrict__ bet,
               float* __restrict__ O, __bf16* __restrict__ Obf)
{
    __shared__ float red[256];
    __shared__ float mean_s, rstd_s;
    const int row = blockIdx.x;
    const int t   = threadIdx.x;
    const float* xr = X + (size_t)row * Hc;
    const bool hasPos = (pos != nullptr);
    const float* pr = hasPos ? pos + (size_t)(row % Nseq) * Hc : nullptr;

    float vals[3];
    float sum = 0.f;
#pragma unroll
    for (int i = 0; i < 3; ++i) {
        int c = t + i * 256;
        float v = xr[c];
        if (hasPos) v += pr[c];
        vals[i] = v; sum += v;
    }
    red[t] = sum; __syncthreads();
    for (int s = 128; s > 0; s >>= 1) { if (t < s) red[t] += red[t + s]; __syncthreads(); }
    if (t == 0) mean_s = red[0] * (1.0f / Hc);
    __syncthreads();
    const float m = mean_s;
    float vs = 0.f;
#pragma unroll
    for (int i = 0; i < 3; ++i) { float d = vals[i] - m; vs += d * d; }
    red[t] = vs; __syncthreads();
    for (int s = 128; s > 0; s >>= 1) { if (t < s) red[t] += red[t + s]; __syncthreads(); }
    if (t == 0) rstd_s = rsqrtf(red[0] * (1.0f / Hc) + 1e-12f);
    __syncthreads();
    const float rs = rstd_s;
#pragma unroll
    for (int i = 0; i < 3; ++i) {
        int c = t + i * 256;
        float o = (vals[i] - m) * rs * gam[c] + bet[c];
        O[(size_t)row * Hc + c] = o;
        if (Obf) Obf[(size_t)row * Hc + c] = (__bf16)o;
    }
}

// ---------------------------------------------------------------------------
// scores S[(b*24+s)][n] = (X[b,n,:] @ Wa[:,s] + ba[s]) / 8 + ext[b,n]
// ---------------------------------------------------------------------------
__global__ __launch_bounds__(256)
void score_kernel(const float* __restrict__ X, const float* __restrict__ Wa,
                  const float* __restrict__ ba, const float* __restrict__ mask,
                  float* __restrict__ S)
{
    const int idx = blockIdx.x * 256 + threadIdx.x;     // over B*N*24
    if (idx >= Mrows * HG) return;
    const int s  = idx % HG;
    const int rn = idx / HG;
    const int b  = rn / Nseq;
    const int n  = rn % Nseq;
    const float* xr = X + (size_t)rn * Hc;
    float acc = ba[s];
    for (int k = 0; k < Hc; ++k) acc += xr[k] * Wa[(size_t)k * HG + s];
    acc *= 0.125f;
    acc += (1.0f - mask[rn]) * -10000.0f;
    S[((size_t)b * HG + s) * Nseq + n] = acc;
}

__global__ __launch_bounds__(256)
void softmax_rows(float* __restrict__ S)
{
    __shared__ float red[256];
    float* row = S + (size_t)blockIdx.x * Nseq;
    const int t = threadIdx.x;
    float m = -1e30f;
    for (int n = t; n < Nseq; n += 256) m = fmaxf(m, row[n]);
    red[t] = m; __syncthreads();
    for (int s = 128; s > 0; s >>= 1) { if (t < s) red[t] = fmaxf(red[t], red[t + s]); __syncthreads(); }
    m = red[0]; __syncthreads();
    float sum = 0.f;
    for (int n = t; n < Nseq; n += 256) { float e = __expf(row[n] - m); row[n] = e; sum += e; }
    red[t] = sum; __syncthreads();
    for (int s = 128; s > 0; s >>= 1) { if (t < s) red[t] += red[t + s]; __syncthreads(); }
    const float inv = 1.0f / red[0];
    __syncthreads();
    for (int n = t; n < Nseq; n += 256) row[n] *= inv;
}

// G[(b*24 + h*2 + g)][d] = sum_n S[(b*24+h*2+g)][n] * X[b,n,h*64+d]
__global__ __launch_bounds__(64)
void globvec_kernel(const float* __restrict__ S, const float* __restrict__ X,
                    float* __restrict__ G)
{
    const int bs = blockIdx.x;
    const int b  = bs / HG;
    const int h  = (bs % HG) >> 1;
    const int d  = threadIdx.x;
    const float* srow = S + (size_t)bs * Nseq;
    float acc = 0.f;
    for (int n = 0; n < Nseq; ++n)
        acc += srow[n] * X[((size_t)b * Nseq + n) * Hc + h * Dd + d];
    G[(size_t)bs * Dd + d] = acc;
}

// Out[b,n,c] = max_g X[b,n,c] * G[b,h(c),g,d(c)]  (+ optional bf16 mirror)
__global__ __launch_bounds__(256)
void maxmul_kernel(const float* __restrict__ X, const float* __restrict__ G,
                   float* __restrict__ Out, __bf16* __restrict__ Obf)
{
    const size_t idx = (size_t)blockIdx.x * 256 + threadIdx.x;  // B*N*H
    if (idx >= (size_t)Mrows * Hc) return;
    const int c = (int)(idx % Hc);
    const int b = (int)((idx / Hc) / Nseq);
    const int h = c >> 6;
    const int d = c & 63;
    const float x  = X[idx];
    const float g0 = G[((size_t)(b * HG + h * 2 + 0)) * Dd + d];
    const float g1 = G[((size_t)(b * HG + h * 2 + 1)) * Dd + d];
    const float r  = fmaxf(x * g0, x * g1);
    Out[idx] = r;
    if (Obf) Obf[idx] = (__bf16)r;
}

// alpha[b*N+n] = exp(E[b,n,:]@p2w + p2b) * mask[b,n]
__global__ __launch_bounds__(256)
void alpha_kernel(const float* __restrict__ E, const float* __restrict__ p2w,
                  const float* __restrict__ p2b, const float* __restrict__ mask,
                  float* __restrict__ alpha)
{
    __shared__ float red[256];
    const int row = blockIdx.x;
    const int t   = threadIdx.x;
    const float* er = E + (size_t)row * Hc;
    float s = 0.f;
    for (int i = t; i < Hc; i += 256) s += er[i] * p2w[i];
    red[t] = s; __syncthreads();
    for (int st = 128; st > 0; st >>= 1) { if (t < st) red[t] += red[t + st]; __syncthreads(); }
    if (t == 0) alpha[row] = __expf(red[0] + p2b[0]) * mask[row];
}

__global__ __launch_bounds__(256)
void asum_kernel(const float* __restrict__ alpha, float* __restrict__ asum)
{
    __shared__ float red[256];
    const int b = blockIdx.x;
    const int t = threadIdx.x;
    red[t] = alpha[b * Nseq + t] + alpha[b * Nseq + t + 256];
    __syncthreads();
    for (int st = 128; st > 0; st >>= 1) { if (t < st) red[t] += red[t + st]; __syncthreads(); }
    if (t == 0) asum[b] = red[0] + 1e-8f;
}

// out[b, d] = sum_n X[b,n,d] * alpha[b,n] / asum[b]
__global__ __launch_bounds__(256)
void pool_kernel(const float* __restrict__ X, const float* __restrict__ alpha,
                 const float* __restrict__ asum, float* __restrict__ out)
{
    const int b = blockIdx.x;
    const int t = threadIdx.x;
    const float inv = 1.0f / asum[b];
#pragma unroll
    for (int i = 0; i < 3; ++i) {
        const int d = t + i * 256;
        float acc = 0.f;
        for (int n = 0; n < Nseq; ++n)
            acc += X[((size_t)b * Nseq + n) * Hc + d] * alpha[b * Nseq + n];
        out[b * Hc + d] = acc * inv;
    }
}

// ---------------------------------------------------------------------------
extern "C" void kernel_launch(void* const* d_in, const int* in_sizes, int n_in,
                              void* d_out, int out_size, void* d_ws, size_t ws_size,
                              hipStream_t stream)
{
    (void)in_sizes; (void)n_in; (void)out_size; (void)ws_size;
    const float* input_embs = (const float*)d_in[0];
    const float* attn_mask  = (const float*)d_in[1];
    const float* pos_emb    = (const float*)d_in[2];
    const float* ln0_g = (const float*)d_in[3];
    const float* ln0_b = (const float*)d_in[4];
    const float* Wq  = (const float*)d_in[5];
    const float* bq  = (const float*)d_in[6];
    const float* Wqa = (const float*)d_in[7];
    const float* bqa = (const float*)d_in[8];
    const float* Wk  = (const float*)d_in[9];
    const float* bk  = (const float*)d_in[10];
    const float* Wka = (const float*)d_in[11];
    const float* bka = (const float*)d_in[12];
    const float* Wv  = (const float*)d_in[13];
    const float* bv  = (const float*)d_in[14];
    const float* Wt  = (const float*)d_in[15];
    const float* bt  = (const float*)d_in[16];
    const float* Wso = (const float*)d_in[17];
    const float* bso = (const float*)d_in[18];
    const float* ln1_g = (const float*)d_in[19];
    const float* ln1_b = (const float*)d_in[20];
    const float* Wi  = (const float*)d_in[21];
    const float* bi  = (const float*)d_in[22];
    const float* Wo2 = (const float*)d_in[23];
    const float* bo2 = (const float*)d_in[24];
    const float* ln2_g = (const float*)d_in[25];
    const float* ln2_b = (const float*)d_in[26];
    const float* p1w = (const float*)d_in[27];
    const float* p1b = (const float*)d_in[28];
    const float* p2w = (const float*)d_in[29];
    const float* p2b = (const float*)d_in[30];

    // ---- workspace carve-out ----
    char* base = (char*)d_ws;
    size_t off = 0;
    auto alloc = [&](size_t bytes) -> void* {
        void* p = base + off;
        off += (bytes + 255) & ~(size_t)255;
        return p;
    };
    const size_t MH = (size_t)Mrows * Hc;
    float*  xb  = (float*) alloc(MH * 4);
    float*  qb  = (float*) alloc(MH * 4);
    float*  kb  = (float*) alloc(MH * 4);
    float*  vb  = (float*) alloc(MH * 4);
    float*  pb  = (float*) alloc(MH * 4);
    float*  fb  = (float*) alloc((size_t)Mrows * Ic * 4);
    float*  Sb  = (float*) alloc((size_t)Bc * HG * Nseq * 4);
    float*  GQ  = (float*) alloc((size_t)Bc * HG * Dd * 4);
    float*  GK  = (float*) alloc((size_t)Bc * HG * Dd * 4);
    float*  al  = (float*) alloc(Mrows * 4);
    float*  as_ = (float*) alloc(Bc * 4);
    __bf16* xbf = (__bf16*)alloc(MH * 2);
    __bf16* ubf = (__bf16*)alloc(MH * 2);
    __bf16* rbf = (__bf16*)alloc(MH * 2);
    __bf16* abf = (__bf16*)alloc(MH * 2);
    __bf16* fbf = (__bf16*)alloc((size_t)Mrows * Ic * 2);
    __bf16* wscr= (__bf16*)alloc((size_t)Hc * Ic * 2);

    const dim3 gblk(128);                                // 4 waves
    const dim3 gH(Hc / BN, Mrows / BM);                  // (6, 32)
    const dim3 gI(Ic / BN, Mrows / BM);                  // (24, 32)
    const dim3 cblk(256);
    const dim3 cHH(Hc / 32, Hc / 32);                    // 768x768 transpose
    const dim3 cHI(Ic / 32, Hc / 32);                    // Wi: K=Hc, N=Ic
    const dim3 cIH(Hc / 32, Ic / 32);                    // Wo2: K=Ic, N=Hc
    const int scoreBlocks = (Mrows * HG) / 256;
    const int ewBlocks    = (int)((size_t)Mrows * Hc / 256);

    ln_kernel<<<Mrows, 256, 0, stream>>>(input_embs, pos_emb, ln0_g, ln0_b, xb, xbf);

    for (int l = 0; l < Lc; ++l) {
        const float* Wq_l  = Wq  + (size_t)l * Hc * Hc;  const float* bq_l  = bq  + l * Hc;
        const float* Wk_l  = Wk  + (size_t)l * Hc * Hc;  const float* bk_l  = bk  + l * Hc;
        const float* Wv_l  = Wv  + (size_t)l * Hc * Hc;  const float* bv_l  = bv  + l * Hc;
        const float* Wqa_l = Wqa + (size_t)l * Hc * HG;  const float* bqa_l = bqa + l * HG;
        const float* Wka_l = Wka + (size_t)l * Hc * HG;  const float* bka_l = bka + l * HG;
        const float* Wt_l  = Wt  + (size_t)l * Hc * Hc;  const float* bt_l  = bt  + l * Hc;
        const float* Wso_l = Wso + (size_t)l * Hc * Hc;  const float* bso_l = bso + l * Hc;
        const float* Wi_l  = Wi  + (size_t)l * Hc * Ic;  const float* bi_l  = bi  + l * Ic;
        const float* Wo2_l = Wo2 + (size_t)l * Ic * Hc;  const float* bo2_l = bo2 + l * Hc;

        cvt_w_kernel<<<cHH, cblk, 0, stream>>>(Wq_l, wscr, Hc, Hc);
        gemm_wmma<0,false,Hc><<<gH, gblk, 0, stream>>>(xbf, wscr, bq_l, nullptr, qb, nullptr, Hc);
        cvt_w_kernel<<<cHH, cblk, 0, stream>>>(Wk_l, wscr, Hc, Hc);
        gemm_wmma<0,false,Hc><<<gH, gblk, 0, stream>>>(xbf, wscr, bk_l, nullptr, kb, nullptr, Hc);
        cvt_w_kernel<<<cHH, cblk, 0, stream>>>(Wv_l, wscr, Hc, Hc);
        gemm_wmma<0,false,Hc><<<gH, gblk, 0, stream>>>(xbf, wscr, bv_l, nullptr, vb, nullptr, Hc);

        // global-query weights: scores -> softmax over sequence
        score_kernel<<<scoreBlocks, 256, 0, stream>>>(qb, Wqa_l, bqa_l, attn_mask, Sb);
        softmax_rows<<<Bc * HG, 256, 0, stream>>>(Sb);
        globvec_kernel<<<Bc * HG, 64, 0, stream>>>(Sb, qb, GQ);
        maxmul_kernel<<<ewBlocks, 256, 0, stream>>>(kb, GQ, pb, nullptr);        // qk_flat

        // key scores (NO softmax, faithful) -> global key -> u
        score_kernel<<<scoreBlocks, 256, 0, stream>>>(pb, Wka_l, bka_l, attn_mask, Sb);
        globvec_kernel<<<Bc * HG, 64, 0, stream>>>(Sb, kb, GK);
        maxmul_kernel<<<ewBlocks, 256, 0, stream>>>(vb, GK, vb, ubf);            // u

        // r = u@Wt + bt + q ; a = LN(r@Wso + bso + x)
        cvt_w_kernel<<<cHH, cblk, 0, stream>>>(Wt_l, wscr, Hc, Hc);
        gemm_wmma<1,true ,Hc><<<gH, gblk, 0, stream>>>(ubf, wscr, bt_l, qb, pb, rbf, Hc);
        cvt_w_kernel<<<cHH, cblk, 0, stream>>>(Wso_l, wscr, Hc, Hc);
        gemm_wmma<1,false,Hc><<<gH, gblk, 0, stream>>>(rbf, wscr, bso_l, xb, kb, nullptr, Hc);
        ln_kernel<<<Mrows, 256, 0, stream>>>(kb, nullptr, ln1_g + l * Hc, ln1_b + l * Hc, qb, abf);

        // FFN
        cvt_w_kernel<<<cHI, cblk, 0, stream>>>(Wi_l, wscr, Hc, Ic);
        gemm_wmma<2,true ,Hc><<<gI, gblk, 0, stream>>>(abf, wscr, bi_l, nullptr, fb, fbf, Ic);
        cvt_w_kernel<<<cIH, cblk, 0, stream>>>(Wo2_l, wscr, Ic, Hc);
        gemm_wmma<1,false,Ic><<<gH, gblk, 0, stream>>>(fbf, wscr, bo2_l, qb, kb, nullptr, Hc);
        ln_kernel<<<Mrows, 256, 0, stream>>>(kb, nullptr, ln2_g + l * Hc, ln2_b + l * Hc, xb, xbf);
    }

    // Attention pooling
    cvt_w_kernel<<<cHH, cblk, 0, stream>>>(p1w, wscr, Hc, Hc);
    gemm_wmma<3,false,Hc><<<gH, gblk, 0, stream>>>(xbf, wscr, p1b, nullptr, qb, nullptr, Hc);
    alpha_kernel<<<Mrows, 256, 0, stream>>>(qb, p2w, p2b, attn_mask, al);
    asum_kernel<<<Bc, 256, 0, stream>>>(al, as_);
    pool_kernel<<<Bc, 256, 0, stream>>>(xb, al, as_, (float*)d_out);
}
